// CausalMambaBlock_43250320670963
// MI455X (gfx1250) — compile-verified, matrix operand order checked by reading, not run on hardware
//
#include <hip/hip_runtime.h>

// ---------------- WMMA types ----------------
typedef __attribute__((ext_vector_type(16))) _Float16 v16h;
typedef __attribute__((ext_vector_type(8)))  float    v8f;

#define D_MODEL 1024
#define D_STATE 16
#define D_INNER 2048
#define DT_RANK 64
#define B_SZ 4
#define SEQ_L 2048
#define BL (B_SZ * SEQ_L)   // 8192 rows for all GEMMs
#define NCHUNK 32
#define CH_T 64             // SEQ_L / NCHUNK

// Load a 16x32 fp32 tile (rows along lanes, K along regs) as f16 hi/lo split.
// Layout per CDNA5 ISA 16-bit A-matrix 16x32: lane r (0..15) group kg=lane>>4:
//   halfs 0..7  <- K = k0 + kg*8 + {0..7}
//   halfs 8..15 <- K = k0 + 16 + kg*8 + {0..7}
__device__ __forceinline__ void load_tile_hilo(const float* __restrict__ base,
                                               long ld, int row0, int maxrow,
                                               int k0, v16h& hi, v16h& lo) {
  int lane = threadIdx.x & 31;
  int r = lane & 15;
  int kg = lane >> 4;
  int row = row0 + r;
  if (row > maxrow) row = maxrow;               // clamp; masked at store
  const float* p = base + (long)row * ld + k0 + kg * 8;
#pragma unroll
  for (int c = 0; c < 2; ++c) {
    const float4* q = (const float4*)(p + c * 16);
    float4 f0 = q[0];
    float4 f1 = q[1];
    float fv[8] = {f0.x, f0.y, f0.z, f0.w, f1.x, f1.y, f1.z, f1.w};
#pragma unroll
    for (int j = 0; j < 8; ++j) {
      float f = fv[j];
      _Float16 h = (_Float16)f;
      hi[c * 8 + j] = h;
      lo[c * 8 + j] = (_Float16)(f - (float)h);
    }
  }
}

__device__ __forceinline__ float softplus_f(float x) {
  return fmaxf(x, 0.f) + log1pf(expf(-fabsf(x)));   // stable softplus
}
__device__ __forceinline__ float silu_f(float x) {
  return x / (1.f + expf(-x));
}

// ---------------- WMMA GEMM: C = epi(A(M,K) @ W(N,K)^T + bias) ----------------
// EPI: 0 = none, 1 = softplus. Columns < split go to C1, the rest to C2.
// Block: 256 threads = 8 waves (2 x 4), wave tile 32x32, block tile 64x128.
template <int EPI>
__global__ __launch_bounds__(256) void wmma_gemm_kernel(
    const float* __restrict__ A, long lda,
    const float* __restrict__ W, long ldw,
    float* __restrict__ C1, long ldc1,
    float* __restrict__ C2, long ldc2, int split,
    const float* __restrict__ bias,
    int M, int N, int K) {
  int wave = threadIdx.x >> 5;
  int wm = wave >> 2;        // 0..1
  int wn = wave & 3;         // 0..3
  int m0 = blockIdx.y * 64 + wm * 32;
  int n0 = blockIdx.x * 128 + wn * 32;
  if (m0 >= M || n0 >= N) return;   // uniform per wave -> whole wave exits

  v8f acc[2][2] = {};

  for (int k0 = 0; k0 < K; k0 += 32) {
    v16h ah[2], al[2], bh[2], bl[2];
    load_tile_hilo(A, lda, m0,      M - 1, k0, ah[0], al[0]);
    load_tile_hilo(A, lda, m0 + 16, M - 1, k0, ah[1], al[1]);
    load_tile_hilo(W, ldw, n0,      N - 1, k0, bh[0], bl[0]);
    load_tile_hilo(W, ldw, n0 + 16, N - 1, k0, bh[1], bl[1]);
#pragma unroll
    for (int i = 0; i < 2; ++i) {
#pragma unroll
      for (int j = 0; j < 2; ++j) {
        // fp32-accuracy split product: hi*hi + hi*lo + lo*hi
        acc[i][j] = __builtin_amdgcn_wmma_f32_16x16x32_f16(
            false, ah[i], false, bh[j], (short)0, acc[i][j], false, false);
        acc[i][j] = __builtin_amdgcn_wmma_f32_16x16x32_f16(
            false, ah[i], false, bl[j], (short)0, acc[i][j], false, false);
        acc[i][j] = __builtin_amdgcn_wmma_f32_16x16x32_f16(
            false, al[i], false, bh[j], (short)0, acc[i][j], false, false);
      }
    }
  }

  // Epilogue. C/D layout: VGPR v -> row v + 8*(lane>=16); col = lane & 15.
  int lane = threadIdx.x & 31;
  int cn = lane & 15;
  int hl = lane >> 4;
#pragma unroll
  for (int i = 0; i < 2; ++i) {
#pragma unroll
    for (int j = 0; j < 2; ++j) {
      int col = n0 + j * 16 + cn;
#pragma unroll
      for (int v = 0; v < 8; ++v) {
        int row = m0 + i * 16 + v + hl * 8;
        if (row < M && col < N) {
          float x = acc[i][j][v];
          if (bias) x += bias[col];
          if (EPI == 1) x = softplus_f(x);
          if (col < split) C1[(long)row * ldc1 + col] = x;
          else             C2[(long)row * ldc2 + (col - split)] = x;
        }
      }
    }
  }
}

// ---------------- causal depthwise conv1d (K=4) + bias + SiLU ----------------
__global__ __launch_bounds__(256) void conv_silu_kernel(
    const float* __restrict__ xin,    // (B, L, D)
    const float* __restrict__ cw,     // (D, 4)
    const float* __restrict__ cb,     // (D)
    float* __restrict__ xc,           // (B, L, D)
    int L, int D) {
  long idx = (long)blockIdx.x * blockDim.x + threadIdx.x;
  long total = (long)B_SZ * L * D;
  if (idx >= total) return;
  int d = (int)(idx % D);
  int l = (int)((idx / D) % L);
  int b = (int)(idx / ((long)D * L));
  const float* base = xin + (long)b * L * D + d;
  float acc = cb[d];
#pragma unroll
  for (int k = 0; k < 4; ++k) {
    int t = l - 3 + k;                 // w[d,3] multiplies x[t] (causal)
    if (t >= 0) acc = fmaf(cw[d * 4 + k], base[(long)t * D], acc);
  }
  xc[idx] = silu_f(acc);
}

// ---------------- chunked selective scan ----------------
// h_t = exp(dt_t*A)*h_{t-1} + dt_t*x_t*B_t is a linear recurrence; split L into
// NCHUNK chunks of CH_T, compute local (h_end, P=exp(A*sum dt)) in parallel,
// compose carries sequentially (cheap), then replay with correct h_init.

// Pass A: per (b, chunk, d): local recurrence from h=0; store P[n], h_end[n].
__global__ __launch_bounds__(256) void scan_passA_kernel(
    const float* __restrict__ xc,     // (B, L, D)
    const float* __restrict__ dt,     // (B, L, D)
    const float* __restrict__ xp,     // (B, L, 96); cols 64..79 = B_t
    const float* __restrict__ A_log,  // (D, 16)
    float* __restrict__ bufP,         // (B, NCHUNK, 16, D)
    float* __restrict__ bufH,         // (B, NCHUNK, 16, D)  h_end
    int L, int D) {
  __shared__ float sB[CH_T][D_STATE];
  int b = blockIdx.y;
  int c = blockIdx.z;
  int d = blockIdx.x * 256 + threadIdx.x;
  int t0 = c * CH_T;

  float Aneg[D_STATE];
#pragma unroll
  for (int n = 0; n < D_STATE; ++n) Aneg[n] = -expf(A_log[(long)d * D_STATE + n]);

  for (int i = threadIdx.x; i < CH_T * D_STATE; i += 256) {
    int tt = i >> 4;
    int n = i & 15;
    sB[tt][n] = xp[((long)b * L + t0 + tt) * 96 + 64 + n];
  }
  __syncthreads();

  float h[D_STATE];
#pragma unroll
  for (int n = 0; n < D_STATE; ++n) h[n] = 0.f;
  float sdt = 0.f;

  for (int tt = 0; tt < CH_T; ++tt) {
    long off = ((long)b * L + t0 + tt) * D + d;
    float xv = xc[off];
    float dtv = dt[off];
    float dtx = dtv * xv;
    sdt += dtv;
#pragma unroll
    for (int n = 0; n < D_STATE; ++n) {
      float dA = expf(dtv * Aneg[n]);
      h[n] = fmaf(dA, h[n], dtx * sB[tt][n]);
    }
  }

  long so = (((long)b * NCHUNK + c) * D_STATE) * D + d;
#pragma unroll
  for (int n = 0; n < D_STATE; ++n) {
    bufP[so + (long)n * D] = expf(sdt * Aneg[n]);  // product of chunk decays
    bufH[so + (long)n * D] = h[n];
  }
}

// Carry pass: per (b, d): compose chunks sequentially; bufH becomes h_init.
__global__ __launch_bounds__(256) void scan_carry_kernel(
    const float* __restrict__ bufP,
    float* __restrict__ bufH,         // in: h_end, out: h_init per chunk
    int D) {
  int idx = blockIdx.x * 256 + threadIdx.x;   // over B*D
  int b = idx / D;
  int d = idx % D;
  float carry[D_STATE];
#pragma unroll
  for (int n = 0; n < D_STATE; ++n) carry[n] = 0.f;
  for (int c = 0; c < NCHUNK; ++c) {
    long so = (((long)b * NCHUNK + c) * D_STATE) * D + d;
#pragma unroll
    for (int n = 0; n < D_STATE; ++n) {
      long o = so + (long)n * D;
      float P  = bufP[o];
      float he = bufH[o];
      bufH[o] = carry[n];                       // h_init for chunk c
      carry[n] = fmaf(P, carry[n], he);
    }
  }
}

// Pass C: replay chunk from h_init; y = (h.C + Dp*x) * silu(z).
__global__ __launch_bounds__(256) void scan_passC_kernel(
    const float* __restrict__ xc,     // (B, L, D)
    const float* __restrict__ dt,     // (B, L, D)
    const float* __restrict__ xp,     // (B, L, 96); B at +64, C at +80
    const float* __restrict__ A_log,  // (D, 16)
    const float* __restrict__ Dp,     // (D)
    const float* __restrict__ z,      // (B, L, D)
    const float* __restrict__ bufH,   // h_init per (b, chunk, n, d)
    float* __restrict__ y,            // (B, L, D)
    int L, int D) {
  __shared__ float sB[CH_T][D_STATE];
  __shared__ float sC[CH_T][D_STATE];
  int b = blockIdx.y;
  int c = blockIdx.z;
  int d = blockIdx.x * 256 + threadIdx.x;
  int t0 = c * CH_T;

  float Aneg[D_STATE];
#pragma unroll
  for (int n = 0; n < D_STATE; ++n) Aneg[n] = -expf(A_log[(long)d * D_STATE + n]);
  float Dpd = Dp[d];

  for (int i = threadIdx.x; i < CH_T * 32; i += 256) {
    int tt = i >> 5;
    int k = i & 31;
    float v = xp[((long)b * L + t0 + tt) * 96 + 64 + k];
    if (k < D_STATE) sB[tt][k] = v;
    else             sC[tt][k - D_STATE] = v;
  }
  __syncthreads();

  float h[D_STATE];
  long so = (((long)b * NCHUNK + c) * D_STATE) * D + d;
#pragma unroll
  for (int n = 0; n < D_STATE; ++n) h[n] = bufH[so + (long)n * D];

  for (int tt = 0; tt < CH_T; ++tt) {
    long off = ((long)b * L + t0 + tt) * D + d;
    float xv = xc[off];
    float dtv = dt[off];
    float dtx = dtv * xv;
    float yv = 0.f;
#pragma unroll
    for (int n = 0; n < D_STATE; ++n) {
      float dA = expf(dtv * Aneg[n]);
      h[n] = fmaf(dA, h[n], dtx * sB[tt][n]);
      yv = fmaf(h[n], sC[tt][n], yv);
    }
    yv = fmaf(Dpd, xv, yv);
    float zv = z[off];
    y[off] = yv * silu_f(zv);
  }
}

// ---------------- launcher ----------------
extern "C" void kernel_launch(void* const* d_in, const int* in_sizes, int n_in,
                              void* d_out, int out_size, void* d_ws, size_t ws_size,
                              hipStream_t stream) {
  const float* u      = (const float*)d_in[0];  // (B, L, 1024)
  const float* W_in   = (const float*)d_in[1];  // (4096, 1024)
  const float* conv_w = (const float*)d_in[2];  // (2048, 4)
  const float* conv_b = (const float*)d_in[3];  // (2048)
  const float* W_x    = (const float*)d_in[4];  // (96, 2048)
  const float* W_dt   = (const float*)d_in[5];  // (2048, 64)
  const float* b_dt   = (const float*)d_in[6];  // (2048)
  const float* A_log  = (const float*)d_in[7];  // (2048, 16)
  const float* Dp     = (const float*)d_in[8];  // (2048)
  const float* W_out  = (const float*)d_in[9];  // (1024, 2048)
  float* out = (float*)d_out;                   // (B, L, 1024) fp32

  float* ws = (float*)d_ws;
  const size_t MN = (size_t)BL * D_INNER;                       // 16,777,216
  const size_t ST = (size_t)B_SZ * NCHUNK * D_STATE * D_INNER;  // 4,194,304
  float* bufX  = ws;            // x_in, reused as y after the scan
  float* bufZ  = ws + MN;       // z
  float* bufXC = ws + 2 * MN;   // conv+silu output
  float* bufDT = ws + 3 * MN;   // softplus dt, (BL, 2048)
  float* bufXP = ws + 4 * MN;   // (BL, 96)
  float* bufP  = bufXP + (size_t)BL * 96;   // chunk decay products
  float* bufH  = bufP + ST;                 // chunk h_end -> h_init

  dim3 blk(256);

  // 1) xz = u @ W_in^T ; split columns into x_in | z
  wmma_gemm_kernel<0><<<dim3((2 * D_INNER) / 128, BL / 64), blk, 0, stream>>>(
      u, D_MODEL, W_in, D_MODEL,
      bufX, D_INNER, bufZ, D_INNER, D_INNER, nullptr,
      BL, 2 * D_INNER, D_MODEL);

  // 2) causal depthwise conv + SiLU
  conv_silu_kernel<<<(unsigned)((MN + 255) / 256), blk, 0, stream>>>(
      bufX, conv_w, conv_b, bufXC, SEQ_L, D_INNER);

  // 3) xp = x @ W_x^T  (N = 96)
  wmma_gemm_kernel<0><<<dim3(1, BL / 64), blk, 0, stream>>>(
      bufXC, D_INNER, W_x, D_INNER,
      bufXP, 96, bufXP, 96, 96, nullptr,
      BL, DT_RANK + 2 * D_STATE, D_INNER);

  // 4) dt = softplus(xp[:, :64] @ W_dt^T + b_dt)
  wmma_gemm_kernel<1><<<dim3(D_INNER / 128, BL / 64), blk, 0, stream>>>(
      bufXP, 96, W_dt, DT_RANK,
      bufDT, D_INNER, bufDT, D_INNER, D_INNER, b_dt,
      BL, D_INNER, DT_RANK);

  // 5) chunked selective scan (A: local chunks, carry, C: replay + epilogue)
  scan_passA_kernel<<<dim3(D_INNER / 256, B_SZ, NCHUNK), blk, 0, stream>>>(
      bufXC, bufDT, bufXP, A_log, bufP, bufH, SEQ_L, D_INNER);
  scan_carry_kernel<<<(B_SZ * D_INNER) / 256, blk, 0, stream>>>(
      bufP, bufH, D_INNER);
  scan_passC_kernel<<<dim3(D_INNER / 256, B_SZ, NCHUNK), blk, 0, stream>>>(
      bufXC, bufDT, bufXP, A_log, Dp, bufZ, bufH, bufX, SEQ_L, D_INNER);

  // 6) out = y @ W_out^T
  wmma_gemm_kernel<0><<<dim3(D_MODEL / 128, BL / 64), blk, 0, stream>>>(
      bufX, D_INNER, W_out, D_INNER,
      out, D_MODEL, out, D_MODEL, D_MODEL, nullptr,
      BL, D_MODEL, D_INNER);
}
